// PointNetSetAbstraction_68496138437063
// MI455X (gfx1250) — matrix operand order repaired and checked by reading, not compiled, not run on hardware
//
#include <hip/hip_runtime.h>

// ---------------------------------------------------------------------------
// PointNet SA (group_all) for MI455X / gfx1250.
// GEMMs run on v_wmma_f32_16x16x32_bf16; activations live in bf16 packed in
// the exact WMMA B-fragment layout so GEMM loads are 32B/lane contiguous.
// Wave tile 64x64 -> 16 WMMAs per 8KB of fragment loads (~33 FLOP/byte).
// ---------------------------------------------------------------------------

#define BN_EPS 1e-5f
#define NPTS   2048          // points per batch
#define NC     32768         // B * NPTS = 16 * 2048 columns
#define NTILES 2048          // NC / 16

typedef __attribute__((ext_vector_type(16))) __bf16          bf16x16;
typedef __attribute__((ext_vector_type(16))) unsigned short  u16x16;
typedef __attribute__((ext_vector_type(8)))  float           f32x8;

union FragU { u16x16 u; bf16x16 b; };

__device__ __forceinline__ unsigned short f32_to_bf16(float f) {
    union { float f; unsigned u; } v; v.f = f;
    unsigned r = v.u + 0x7FFFu + ((v.u >> 16) & 1u);   // round-to-nearest-even
    return (unsigned short)(r >> 16);
}
__device__ __forceinline__ float bf16_to_f32(unsigned short h) {
    union { unsigned u; float f; } v; v.u = ((unsigned)h) << 16;
    return v.f;
}

// ---------------------------------------------------------------------------
// Pack weights W[M x Kreal] (f32 row-major) -> A-fragment layout, zero-pad K.
// A 16x32 bf16 frag: lane l holds row m = mt*16 + (l&15);
// element e -> k = e + 16*(l>=16)   (contiguous, matches B-fragment layout).
// ---------------------------------------------------------------------------
__global__ __launch_bounds__(256)
void pack_weights_kernel(const float* __restrict__ W, unsigned short* __restrict__ Wp,
                         int ktiles, int Kreal, int mtiles)
{
    int idx = blockIdx.x * blockDim.x + threadIdx.x;
    int total = mtiles * ktiles * 32;
    if (idx >= total) return;
    int lane = idx & 31;
    int frag = idx >> 5;
    int kt = frag % ktiles;
    int mt = frag / ktiles;
    int lo = lane & 15, h = lane >> 4;
    int m = mt * 16 + lo;
    u16x16 v;
#pragma unroll
    for (int e = 0; e < 16; ++e) {
        int k = kt * 32 + e + h * 16;
        float x = (k < Kreal) ? W[(size_t)m * Kreal + k] : 0.0f;
        v[e] = f32_to_bf16(x);
    }
    *(u16x16*)(Wp + (size_t)frag * 512 + (size_t)lane * 16) = v;
}

// ---------------------------------------------------------------------------
// Pack layer-1 input: rows 0..2 = xyz[b, k, n], rows 3..258 = pts[b, k-3, n],
// rows >= Kreal zero.  B 32x16 bf16 frag: lane l holds col n = nt*16 + (l&15);
// element e -> k = e + 16*(l>=16).
// ---------------------------------------------------------------------------
__global__ __launch_bounds__(256)
void pack_input_kernel(const float* __restrict__ xyz, const float* __restrict__ pts,
                       unsigned short* __restrict__ Xp, int ktiles, int Kreal, int D)
{
    int idx = blockIdx.x * blockDim.x + threadIdx.x;
    int total = NTILES * ktiles * 32;
    if (idx >= total) return;
    int lane = idx & 31;
    int frag = idx >> 5;
    int kt = frag % ktiles;
    int nt = frag / ktiles;
    int lo = lane & 15, h = lane >> 4;
    int col = nt * 16 + lo;
    int b = col >> 11;            // col / 2048
    int n = col & 2047;
    u16x16 v;
#pragma unroll
    for (int e = 0; e < 16; ++e) {
        int k = kt * 32 + e + h * 16;
        float x = 0.0f;
        if (k < 3)          x = xyz[((size_t)b * 3 + k) * NPTS + n];
        else if (k < Kreal) x = pts[((size_t)b * D + (k - 3)) * NPTS + n];
        v[e] = f32_to_bf16(x);
    }
    *(u16x16*)(Xp + (size_t)frag * 512 + (size_t)lane * 16) = v;
}

// ---------------------------------------------------------------------------
// GEMM: Y[M x NC] = W @ X + bias, bf16 WMMA / f32 accum.
// Wave tile = 64 rows x 64 cols (4 A-frags x 4 B-frags -> 16 WMMAs / k-step).
// Epilogue: per-channel sum / sumsq atomics for BN, then repack D-layout into
// next-layer B-fragment layout (one shfl_xor(16) pair) and store bf16.
// Block = 4 waves; grid = (NC/256, M/64).
// ---------------------------------------------------------------------------
__global__ __launch_bounds__(128)
void gemm_bias_stats_kernel(const unsigned short* __restrict__ Wp,
                            const unsigned short* __restrict__ Xp,
                            unsigned short* __restrict__ Yp,
                            const float* __restrict__ bias,
                            float* __restrict__ sums,
                            float* __restrict__ sumsq,
                            int ktiles, int M)
{
    const int lane = threadIdx.x & 31;
    const int wave = threadIdx.x >> 5;
    const int rg   = blockIdx.y;                 // 64-row group
    const int cg   = blockIdx.x * 4 + wave;      // 64-col group
    const int nt0  = cg * 4;
    const int mt0  = rg * 4;
    const int otiles = M >> 5;                   // k-tiles of the output tensor
    const size_t laneOff = (size_t)lane * 16;

    f32x8 acc[4][4];                             // [m-tile i][n-tile j]
    const f32x8 zero = {0.f, 0.f, 0.f, 0.f, 0.f, 0.f, 0.f, 0.f};
#pragma unroll
    for (int i = 0; i < 4; ++i)
#pragma unroll
        for (int j = 0; j < 4; ++j) acc[i][j] = zero;

    const unsigned short* aBase = Wp + (size_t)mt0 * ktiles * 512 + laneOff;
    const unsigned short* xBase = Xp + (size_t)nt0 * ktiles * 512 + laneOff;

    for (int kt = 0; kt < ktiles; ++kt) {
        FragU afr[4], bfr[4];
#pragma unroll
        for (int i = 0; i < 4; ++i)
            afr[i].u = *(const u16x16*)(aBase + ((size_t)i * ktiles + kt) * 512);
#pragma unroll
        for (int j = 0; j < 4; ++j)
            bfr[j].u = *(const u16x16*)(xBase + ((size_t)j * ktiles + kt) * 512);
#pragma unroll
        for (int i = 0; i < 4; ++i)
#pragma unroll
            for (int j = 0; j < 4; ++j)
                acc[i][j] = __builtin_amdgcn_wmma_f32_16x16x32_bf16(
                    false, afr[i].b, false, bfr[j].b, (short)0, acc[i][j], false, false);
    }

    const int half = lane >> 4;
    const int lo   = lane & 15;

    int rbase[4];
    float bz[4][8];
#pragma unroll
    for (int i = 0; i < 4; ++i) {
        rbase[i] = rg * 64 + i * 16 + half * 8;  // rows of acc[i][.][v]
#pragma unroll
        for (int v = 0; v < 8; ++v) bz[i][v] = bias[rbase[i] + v];
    }

    float s1[4][8], s2[4][8];
#pragma unroll
    for (int i = 0; i < 4; ++i)
#pragma unroll
        for (int v = 0; v < 8; ++v) { s1[i][v] = 0.f; s2[i][v] = 0.f; }

#pragma unroll
    for (int i = 0; i < 4; ++i)
#pragma unroll
        for (int j = 0; j < 4; ++j)
#pragma unroll
            for (int v = 0; v < 8; ++v) {
                float y = acc[i][j][v] + bz[i][v];
                acc[i][j][v] = y;
                s1[i][v] += y;
                s2[i][v] += y * y;
            }

    // reduce across the 16 lanes of each half (masks 1,2,4,8 stay in-half)
#pragma unroll
    for (int m = 1; m < 16; m <<= 1)
#pragma unroll
        for (int i = 0; i < 4; ++i)
#pragma unroll
            for (int v = 0; v < 8; ++v) {
                s1[i][v] += __shfl_xor(s1[i][v], m, 32);
                s2[i][v] += __shfl_xor(s2[i][v], m, 32);
            }
    if (lo == 0) {
#pragma unroll
        for (int i = 0; i < 4; ++i)
#pragma unroll
            for (int v = 0; v < 8; ++v) {
                atomicAdd(&sums[rbase[i] + v],  s1[i][v]);
                atomicAdd(&sumsq[rbase[i] + v], s2[i][v]);
            }
    }

    // D-layout -> B-fragment layout (lane l swaps with l^16), store bf16.
    // Rows rg*64 + p*32 .. +31 come from acc[2p], acc[2p+1] -> out k-tile rg*2+p.
#pragma unroll
    for (int j = 0; j < 4; ++j) {
#pragma unroll
        for (int p = 0; p < 2; ++p) {
            u16x16 t;
#pragma unroll
            for (int e = 0; e < 8; ++e) {
                float a0 = acc[2 * p][j][e];
                float a1 = acc[2 * p + 1][j][e];
                float x0 = __shfl_xor(a0, 16, 32);
                float x1 = __shfl_xor(a1, 16, 32);
                float lov = half ? x1 : a0;          // local k = e
                float hiv = half ? a1 : x0;          // local k = e + 8 -> t[e+8]
                t[e]     = f32_to_bf16(lov);
                t[e + 8] = f32_to_bf16(hiv);
            }
            *(u16x16*)(Yp + ((size_t)(nt0 + j) * otiles + rg * 2 + p) * 512 + laneOff) = t;
        }
    }
}

// ---------------------------------------------------------------------------
// Fold BN stats into per-channel scale/shift.
// ---------------------------------------------------------------------------
__global__ __launch_bounds__(256)
void bn_stats_kernel(const float* __restrict__ sums, const float* __restrict__ sumsq,
                     const float* __restrict__ gamma, const float* __restrict__ beta,
                     float* __restrict__ scale, float* __restrict__ shift, int M)
{
    int c = blockIdx.x * blockDim.x + threadIdx.x;
    if (c >= M) return;
    const float inv = 1.0f / (float)NC;
    float mean = sums[c] * inv;
    float var  = sumsq[c] * inv - mean * mean;   // biased, matches torch BN train
    float sc   = gamma[c] * rsqrtf(var + BN_EPS);
    scale[c] = sc;
    shift[c] = beta[c] - mean * sc;
}

// ---------------------------------------------------------------------------
// In-place relu(scale*y + shift) over the packed bf16 tensor (layout is
// already correct for the next GEMM's B operand).
// ---------------------------------------------------------------------------
__global__ __launch_bounds__(256)
void bn_relu_inplace_kernel(unsigned short* __restrict__ Yp,
                            const float* __restrict__ scale,
                            const float* __restrict__ shift,
                            int ktiles, int total /* frags*32 */)
{
    int idx = blockIdx.x * blockDim.x + threadIdx.x;
    if (idx >= total) return;
    int lane = idx & 31;
    size_t frag = (size_t)(idx >> 5);
    int kt = (int)(frag % (size_t)ktiles);
    int kbase = kt * 32 + ((lane >> 4) << 4);
    unsigned short* p = Yp + frag * 512 + (size_t)lane * 16;
    u16x16 v = *(const u16x16*)p;
#pragma unroll
    for (int e = 0; e < 16; ++e) {
        int k = kbase + e;
        float y = bf16_to_f32(v[e]);
        y = fmaxf(scale[k] * y + shift[k], 0.0f);
        v[e] = f32_to_bf16(y);
    }
    *(u16x16*)p = v;
}

// ---------------------------------------------------------------------------
// out[b*1024 + c] = max_n y[c][(b,n)].  ReLU output >= 0 and d_out is zeroed,
// so monotonic uint atomicMax on the f32 bit pattern is exact.
// ---------------------------------------------------------------------------
__global__ __launch_bounds__(256)
void max_reduce_kernel(const unsigned short* __restrict__ Yp, float* __restrict__ out)
{
    int t = blockIdx.x * blockDim.x + threadIdx.x;   // 16 * 1024 * 8 threads
    if (t >= 16 * 1024 * 8) return;
    int chunk = t & 7;
    int c = (t >> 3) & 1023;
    int b = t >> 13;
    int kt = c >> 5;
    int r = c & 31;
    int h = r >> 4, e = r & 15;
    const int ktiles = 32;                            // M = 1024
    float m = 0.0f;
    int n0 = chunk * 256;
    for (int n = n0; n < n0 + 256; ++n) {
        int col = b * NPTS + n;
        int nt = col >> 4, lo = col & 15;
        unsigned short u = Yp[((size_t)nt * ktiles + kt) * 512 + (size_t)(h * 16 + lo) * 16 + e];
        m = fmaxf(m, bf16_to_f32(u));
    }
    atomicMax((unsigned int*)&out[(size_t)b * 1024 + c], __float_as_uint(m));
}

// ---------------------------------------------------------------------------
// Host orchestration.  Inputs (setup_inputs order):
//   0 xyz  1 points  2 color  3 colors
//   4..15  params_pts  (w,b,gamma,beta) x 3 layers
//   16..27 params_col
// d_out = [new_xyz(48) | new_points(16384) | new_color(48) | new_colors(16384)]
// ---------------------------------------------------------------------------
extern "C" void kernel_launch(void* const* d_in, const int* in_sizes, int n_in,
                              void* d_out, int out_size, void* d_ws, size_t ws_size,
                              hipStream_t stream)
{
    (void)in_sizes; (void)n_in; (void)out_size; (void)ws_size;

    const int Kreal[3] = {259, 256, 512};
    const int ktin[3]  = {9, 8, 16};      // input k-tiles (K padded to mult of 32)
    const int Mout[3]  = {256, 512, 1024};

    char* ws = (char*)d_ws;
    unsigned short* bufA = (unsigned short*)ws;                          // 32 MB
    unsigned short* bufB = (unsigned short*)(ws + ((size_t)32 << 20));   // 64 MB
    unsigned short* Wp0  = (unsigned short*)(ws + ((size_t)96 << 20));
    unsigned short* Wp1  = Wp0 + (size_t)16 * 9 * 512;                   // 144 frags
    unsigned short* Wp2  = Wp1 + (size_t)32 * 8 * 512;                   // 256 frags
    float* sums  = (float*)(ws + ((size_t)98 << 20));
    float* sumsq = sums  + 1024;
    float* scale = sumsq + 1024;
    float* shift = scale + 1024;
    unsigned short* Wpp[3] = {Wp0, Wp1, Wp2};

    // zero all outputs (also produces the zero new_xyz / new_color tensors
    // and the baseline for atomicMax)
    hipMemsetAsync(d_out, 0, (size_t)(48 + 16384 + 48 + 16384) * sizeof(float), stream);

    for (int br = 0; br < 2; ++br) {
        const float* in_xyz = (const float*)d_in[br ? 2 : 0];
        const float* in_pts = (const float*)d_in[br ? 3 : 1];
        const int pb = br ? 16 : 4;
        const float* Wl[3]  = {(const float*)d_in[pb + 0], (const float*)d_in[pb + 4], (const float*)d_in[pb + 8]};
        const float* bl[3]  = {(const float*)d_in[pb + 1], (const float*)d_in[pb + 5], (const float*)d_in[pb + 9]};
        const float* gl[3]  = {(const float*)d_in[pb + 2], (const float*)d_in[pb + 6], (const float*)d_in[pb + 10]};
        const float* bel[3] = {(const float*)d_in[pb + 3], (const float*)d_in[pb + 7], (const float*)d_in[pb + 11]};
        float* outp = (float*)d_out + (br ? (48 + 16384 + 48) : 48);

        // pack weights for all 3 layers of this branch
        for (int l = 0; l < 3; ++l) {
            int mtiles = Mout[l] / 16;
            int tot = mtiles * ktin[l] * 32;
            pack_weights_kernel<<<(tot + 255) / 256, 256, 0, stream>>>(
                Wl[l], Wpp[l], ktin[l], Kreal[l], mtiles);
        }
        // pack concatenated [xyz; points] into bf16 B-fragment layout
        {
            int tot = NTILES * ktin[0] * 32;
            pack_input_kernel<<<(tot + 255) / 256, 256, 0, stream>>>(
                in_xyz, in_pts, bufA, ktin[0], Kreal[0], 256);
        }

        unsigned short* xin = bufA;
        for (int l = 0; l < 3; ++l) {
            unsigned short* xout = (l & 1) ? bufA : bufB;   // Y1->B, Y2->A, Y3->B
            hipMemsetAsync(sums, 0, 2048 * sizeof(float), stream);  // sums+sumsq
            dim3 grid(NC / 256, Mout[l] / 64);
            gemm_bias_stats_kernel<<<grid, 128, 0, stream>>>(
                Wpp[l], xin, xout, bl[l], sums, sumsq, ktin[l], Mout[l]);
            bn_stats_kernel<<<(Mout[l] + 255) / 256, 256, 0, stream>>>(
                sums, sumsq, gl[l], bel[l], scale, shift, Mout[l]);
            int total = NTILES * (Mout[l] >> 5) * 32;
            bn_relu_inplace_kernel<<<(total + 255) / 256, 256, 0, stream>>>(
                xout, scale, shift, Mout[l] >> 5, total);
            xin = xout;
        }
        max_reduce_kernel<<<(16 * 1024 * 8) / 256, 256, 0, stream>>>(xin, outp);
    }
}
